// DKVMNModel_6966436954240
// MI455X (gfx1250) — matrix operand order repaired and checked by reading, not compiled, not run on hardware
//
#include <hip/hip_runtime.h>

#define B_    128
#define S_    1024
#define INDIM 256
#define M_    64
#define D_    128
#define OUT_  128

#define BPB   16              // batches per scan block
#define SSTR  144             // bf16 slab row stride (elements): 256B + 32B TDM pad
#define RSTR  144             // bf16 r-buffer row stride (elements)
#define WSTR  68              // padded stride for w/scores rows (f32)
#define SLAB_B (BPB * SSTR * 2)   // 4608 bytes per bf16 slab

// LDS layout (bytes):
//      0 : s_mv   bf16[16][64][128]            262144
// 262144 : slabs  bf16[2][3][16][144]           27648
// 289792 : s_w    f32[16][68]                    4352
// 294144 : s_r    f32[16][128]  (partial r)      8192
// 302336 : s_rb   bf16[16][144] (bf16 r)     4608
// total  : 306944  (<= 320KB WGP LDS)
#define OFF_SLAB 262144
#define OFF_W    289792
#define OFF_R    294144
#define OFF_RB   302336
#define LDS_TOTAL 306944

typedef __attribute__((ext_vector_type(16))) __bf16         v16bf;
typedef __attribute__((ext_vector_type(16))) unsigned short v16us;
typedef __attribute__((ext_vector_type(8)))  float          v8f;
typedef __attribute__((ext_vector_type(4)))  unsigned int   u32x4;
typedef __attribute__((ext_vector_type(8)))  int            i32x8;
typedef __attribute__((ext_vector_type(4)))  int            i32x4;

__device__ __forceinline__ unsigned short f2bfu(float f) {
  return __builtin_bit_cast(unsigned short, (__bf16)f);   // native cvt, RNE
}
__device__ __forceinline__ float bf2f(unsigned short h) {
  return __uint_as_float(((unsigned)h) << 16);
}
// WMMA-A element-order swizzle of a K/column index: swap bits 3 and 4.
// A contiguous 16-element read at offset 16*half then yields the A-layout
// interleave {8*half+0..7, 16+8*half+0..7}.
__device__ __forceinline__ int swz34(int c) {
  return (c & ~24) | ((c & 8) << 1) | ((c & 16) >> 1);
}

// Build a 16-bit-A WMMA fragment from f32 memory (phase-1 GEMM only):
// elements 0..15 = K(base+16*half+0..15)
__device__ __forceinline__ v16bf make_afrag_f32(const float* p /* at base+16*half */) {
  v16bf a;
  #pragma unroll
  for (int i = 0; i < 8; ++i) { a[i] = (__bf16)p[i]; a[i + 8] = (__bf16)p[8 + i]; }
  return a;
}

// ---------------------------------------------------------------------------
// Kernel 0: transpose + bf16-convert weights (B-operands need column access)
// ---------------------------------------------------------------------------
__global__ void prep_weights(const float* __restrict__ Wr, const float* __restrict__ Ww,
                             const float* __restrict__ Wp,
                             unsigned short* __restrict__ WrT, unsigned short* __restrict__ WwT,
                             unsigned short* __restrict__ WpT) {
  int tid = blockIdx.x * blockDim.x + threadIdx.x;
  if (tid < INDIM * D_) {
    int r = tid / D_, c = tid % D_;
    WrT[c * INDIM + r] = f2bfu(Wr[tid]);
    WwT[c * INDIM + r] = f2bfu(Ww[tid]);
  }
  if (tid < D_ * OUT_) {
    int r = tid / OUT_, c = tid % OUT_;
    WpT[c * D_ + r] = f2bfu(Wp[tid]);
  }
}

// ---------------------------------------------------------------------------
// Kernel 1: k = x@Wr+br (bf16, natural order) ; e = sigmoid(x@Ww+bw) ; a = tanh(...)
// One block = 16 rows of (B*S); 8 waves, wave q owns output n-tile q (N=128).
// A-fragment (x) shared across both GEMMs per k-step.
// ---------------------------------------------------------------------------
__global__ void __launch_bounds__(256)
gemm_act(const float* __restrict__ x,
         const unsigned short* __restrict__ WrT, const unsigned short* __restrict__ WwT,
         const float* __restrict__ br, const float* __restrict__ bw,
         unsigned short* __restrict__ k_bf, unsigned short* __restrict__ e_bf,
         unsigned short* __restrict__ a_bf) {
  const int lane = threadIdx.x & 31;
  const int q    = threadIdx.x >> 5;
  const int lr   = lane & 15;
  const int hl   = lane >> 4;
  const int m0   = blockIdx.x * 16;
  const int n0   = q * 16;

  v8f accK = {}; v8f accV = {};
  #pragma unroll
  for (int ks = 0; ks < 8; ++ks) {
    const float* xp = x + (size_t)(m0 + lr) * INDIM + ks * 32 + 16 * hl;
    v16bf a = make_afrag_f32(xp);
    v16us bu0 = *(const v16us*)(WrT + (size_t)(n0 + lr) * INDIM + ks * 32 + 16 * hl);
    v16us bu1 = *(const v16us*)(WwT + (size_t)(n0 + lr) * INDIM + ks * 32 + 16 * hl);
    v16bf b0 = __builtin_bit_cast(v16bf, bu0);
    v16bf b1 = __builtin_bit_cast(v16bf, bu1);
    accK = __builtin_amdgcn_wmma_f32_16x16x32_bf16(false, a, false, b0, (short)0, accK, false, false);
    accV = __builtin_amdgcn_wmma_f32_16x16x32_bf16(false, a, false, b1, (short)0, accV, false, false);
  }
  const int   nb  = n0 + lr;
  const int   pb  = nb;                 // k is stored in natural column order
  const float brv = br[nb], bwv = bw[nb];
  #pragma unroll
  for (int j = 0; j < 8; ++j) {
    const int    row  = m0 + j + 8 * hl;
    const size_t base = (size_t)row * D_;
    float z = accV[j] + bwv;
    k_bf[base + pb] = f2bfu(accK[j] + brv);
    e_bf[base + nb] = f2bfu(1.0f / (1.0f + __expf(-z)));
    a_bf[base + nb] = f2bfu(tanhf(z));
  }
}

// ---------------------------------------------------------------------------
// TDM: async 2D tile load (16 rows x 128 bf16, row stride S_*D_ elements)
// into LDS with 32B padding per 256B row (32B-aligned, bank-spread rows)
// ---------------------------------------------------------------------------
__device__ __forceinline__ void tdm_load_2d(unsigned lds_addr, const unsigned short* gptr) {
  unsigned long long ga = (unsigned long long)(size_t)gptr;
  u32x4 g0;
  g0[0] = 1u;                                            // count=1, user descriptor
  g0[1] = lds_addr;                                      // LDS dest (bytes)
  g0[2] = (unsigned)(ga & 0xFFFFFFFFu);                  // global_addr[31:0]
  g0[3] = (unsigned)((ga >> 32) & 0x01FFFFFFu) | (2u << 30); // global_addr[56:32] | type=2
  i32x8 g1;
  g1[0] = (1 << 16) | (1 << 20) | (5 << 22) | (7 << 25); // data_size=2B, pad_en, 64dw interval, 8dw pad
  g1[1] = (D_ & 0xFFFF) << 16;                           // tensor_dim0 = 128
  g1[2] = (BPB << 16);                                   // tensor_dim1 = 16
  g1[3] = (D_ << 16);                                    // tile_dim0 = 128
  g1[4] = BPB;                                           // tile_dim1 = 16
  g1[5] = S_ * D_;                                       // tensor_dim0_stride[31:0] (elements)
  g1[6] = 0;
  g1[7] = 0;
  i32x4 z4 = {};
#if __clang_major__ >= 23
  i32x8 z8 = {};
  __builtin_amdgcn_tensor_load_to_lds(g0, g1, z4, z4, z8, 0);
#else
  __builtin_amdgcn_tensor_load_to_lds(g0, g1, z4, z4, 0);
#endif
}

// ---------------------------------------------------------------------------
// Kernel 2: the sequential scan. 8 blocks x 1024 threads (32 waves),
// 16 batch elements per block, Mv resident in LDS (bf16).
// ---------------------------------------------------------------------------
__global__ void __launch_bounds__(1024)
scan_kernel(const unsigned short* __restrict__ k_bf, const unsigned short* __restrict__ e_bf,
            const unsigned short* __restrict__ a_bf, const float* __restrict__ mkey,
            const float* __restrict__ mval0, const unsigned short* __restrict__ WpT,
            const float* __restrict__ bp, float* __restrict__ out) {
  extern __shared__ char smem[];
  unsigned short* s_mv   = (unsigned short*)smem;                // [16][64][128] bf16
  unsigned short* s_slab = (unsigned short*)(smem + OFF_SLAB);   // [2][3][16][144] bf16
  float*          s_w    = (float*)(smem + OFF_W);               // [16][68]
  float*          s_r    = (float*)(smem + OFF_R);               // [16][128] f32 partial
  unsigned short* s_rb   = (unsigned short*)(smem + OFF_RB);     // [16][144] bf16 natural order
  const unsigned  lds_slab = (unsigned)(size_t)(void*)s_slab;

  const int tid  = threadIdx.x;
  const int lane = tid & 31;
  const int wv   = tid >> 5;
  const int lr   = lane & 15;
  const int hl   = lane >> 4;
  const int b0   = blockIdx.x * BPB;

  // ---- preload persistent B-fragments (memory_key for waves 0-3, WpT for 0-7)
  v16bf mkfrag[4] = {};
  v16bf wpfrag[4] = {};
  if (wv < 4) {
    #pragma unroll
    for (int ks = 0; ks < 4; ++ks) {
      const float* p = mkey + (size_t)(wv * 16 + lr) * D_ + ks * 32 + 16 * hl;
      v16bf u;
      #pragma unroll
      for (int i = 0; i < 16; ++i) u[i] = (__bf16)p[i];
      mkfrag[ks] = u;
    }
  }
  if (wv < 8) {
    #pragma unroll
    for (int ks = 0; ks < 4; ++ks) {
      v16us u = *(const v16us*)(WpT + (size_t)(wv * 16 + lr) * D_ + ks * 32 + 16 * hl);
      wpfrag[ks] = __builtin_bit_cast(v16bf, u);
    }
  }
  const float bpv = (wv < 8) ? bp[wv * 16 + lr] : 0.0f;

  // ---- prologue: issue async TDM loads for t=0 (one DMA per array, wave 0)
  if (wv == 0) {
    const size_t g = (size_t)b0 * S_ * D_;
    tdm_load_2d(lds_slab + 0 * SLAB_B, k_bf + g);
    tdm_load_2d(lds_slab + 1 * SLAB_B, e_bf + g);
    tdm_load_2d(lds_slab + 2 * SLAB_B, a_bf + g);
  }

  // ---- init Mv: 16 bf16 copies of memory_value
  #pragma unroll 4
  for (int it = 0; it < (BPB * M_ * D_) / 1024; ++it) {
    int idx = it * 1024 + tid;
    s_mv[idx] = f2bfu(mval0[idx & (M_ * D_ - 1)]);
  }

  const int cb = wv & 15;   // batch owned in phases C/E
  const int mh = wv >> 4;   // m-half owned in phases C/E

  for (int t = 0; t < S_; ++t) {
    const int cur = t & 1;
    if (wv == 0) __builtin_amdgcn_s_wait_tensorcnt(0);   // slab[cur] complete
    __syncthreads();                                     // ...and visible; prev slab free
    if (wv == 0 && t + 1 < S_) {                         // prefetch t+1 into other buffer
      const size_t g = (size_t)b0 * S_ * D_ + (size_t)(t + 1) * D_;
      const unsigned lb = lds_slab + (unsigned)(1 - cur) * (3u * SLAB_B);
      tdm_load_2d(lb + 0 * SLAB_B, k_bf + g);
      tdm_load_2d(lb + 1 * SLAB_B, e_bf + g);
      tdm_load_2d(lb + 2 * SLAB_B, a_bf + g);
    }
    const unsigned short* sk = s_slab + cur * (3 * BPB * SSTR);
    const unsigned short* se = sk + BPB * SSTR;
    const unsigned short* sa = se + BPB * SSTR;

    // ---- A: scores(16x64) = K_t(16x128) @ MKey^T  (waves 0-3, WMMA)
    // k slab is in natural order: raw 32B LDS read == contiguous-k A fragment.
    if (wv < 4) {
      v8f sc = {};
      #pragma unroll
      for (int ks = 0; ks < 4; ++ks) {
        v16us u = *(const v16us*)(sk + lr * SSTR + ks * 32 + 16 * hl);
        v16bf a = __builtin_bit_cast(v16bf, u);
        sc = __builtin_amdgcn_wmma_f32_16x16x32_bf16(false, a, false, mkfrag[ks], (short)0, sc, false, false);
      }
      #pragma unroll
      for (int j = 0; j < 8; ++j)
        s_w[(j + 8 * hl) * WSTR + wv * 16 + lr] = sc[j];
    }
    __syncthreads();

    // ---- B: softmax rows (wave i -> batch i), wave32 butterfly reductions
    if (wv < 16) {
      float v0 = s_w[wv * WSTR + lane];
      float v1 = s_w[wv * WSTR + lane + 32];
      float mx = fmaxf(v0, v1);
      #pragma unroll
      for (int off = 16; off >= 1; off >>= 1) mx = fmaxf(mx, __shfl_xor(mx, off, 32));
      float e0 = __expf(v0 - mx), e1 = __expf(v1 - mx);
      float sm = e0 + e1;
      #pragma unroll
      for (int off = 16; off >= 1; off >>= 1) sm += __shfl_xor(sm, off, 32);
      float inv = 1.0f / sm;
      s_w[wv * WSTR + lane]      = e0 * inv;
      s_w[wv * WSTR + lane + 32] = e1 * inv;
    }
    __syncthreads();

    // ---- C: r[b,:] = w[b,:] @ Mv[b]  (wave pair per batch, split over m)
    float r0 = 0.f, r1 = 0.f, r2 = 0.f, r3 = 0.f;
    {
      const unsigned short* mvp = s_mv + ((size_t)(cb * M_ + mh * 32) * D_ + lane * 4);
      const float* wrow = s_w + cb * WSTR + mh * 32;
      #pragma unroll 4
      for (int mm = 0; mm < 32; ++mm) {
        float wm = wrow[mm];
        uint2 raw = *(const uint2*)(mvp + (size_t)mm * D_);
        r0 += wm * bf2f((unsigned short)(raw.x & 0xFFFF));
        r1 += wm * bf2f((unsigned short)(raw.x >> 16));
        r2 += wm * bf2f((unsigned short)(raw.y & 0xFFFF));
        r3 += wm * bf2f((unsigned short)(raw.y >> 16));
      }
    }
    if (mh == 0) {                      // lower half writes f32 partials
      float* rp = s_r + cb * D_ + lane * 4;
      rp[0] = r0; rp[1] = r1; rp[2] = r2; rp[3] = r3;
    }
    __syncthreads();
    if (mh == 1) {                      // upper half combines + packs natural-order bf16 r
      const float* rp = s_r + cb * D_ + lane * 4;
      uint2 o;
      o.x = (unsigned)f2bfu(rp[0] + r0) | ((unsigned)f2bfu(rp[1] + r1) << 16);
      o.y = (unsigned)f2bfu(rp[2] + r2) | ((unsigned)f2bfu(rp[3] + r3) << 16);
      *(uint2*)(s_rb + cb * RSTR + lane * 4) = o;
    }
    __syncthreads();

    // ---- D: y = sigmoid(r @ Wp + bp)  (waves 0-7, WMMA) -> global out
    // s_rb is in natural order: raw 32B LDS read == contiguous-k A fragment.
    if (wv < 8) {
      v8f acc = {};
      #pragma unroll
      for (int ks = 0; ks < 4; ++ks) {
        v16us u = *(const v16us*)(s_rb + lr * RSTR + ks * 32 + 16 * hl);
        v16bf a = __builtin_bit_cast(v16bf, u);
        acc = __builtin_amdgcn_wmma_f32_16x16x32_bf16(false, a, false, wpfrag[ks], (short)0, acc, false, false);
      }
      #pragma unroll
      for (int j = 0; j < 8; ++j) {
        int bb = j + 8 * hl;
        float y = 1.0f / (1.0f + __expf(-(acc[j] + bpv)));
        out[((size_t)(b0 + bb) * S_ + t) * OUT_ + wv * 16 + lr] = y;
      }
    }

    // ---- E: Mv = Mv*(1 - w*e) + w*a  (all 32 waves; overlaps D on waves 8-31)
    {
      const unsigned short* ep = se + cb * SSTR + lane * 4;
      const unsigned short* ap = sa + cb * SSTR + lane * 4;
      uint2 eraw = *(const uint2*)ep;
      uint2 araw = *(const uint2*)ap;
      float e0 = bf2f((unsigned short)(eraw.x & 0xFFFF));
      float e1 = bf2f((unsigned short)(eraw.x >> 16));
      float e2 = bf2f((unsigned short)(eraw.y & 0xFFFF));
      float e3 = bf2f((unsigned short)(eraw.y >> 16));
      float a0 = bf2f((unsigned short)(araw.x & 0xFFFF));
      float a1 = bf2f((unsigned short)(araw.x >> 16));
      float a2 = bf2f((unsigned short)(araw.y & 0xFFFF));
      float a3 = bf2f((unsigned short)(araw.y >> 16));
      unsigned short* mvp = s_mv + ((size_t)(cb * M_ + mh * 32) * D_ + lane * 4);
      const float* wrow = s_w + cb * WSTR + mh * 32;
      #pragma unroll 2
      for (int mm = 0; mm < 32; ++mm) {
        float wm = wrow[mm];
        uint2 raw = *(uint2*)(mvp + (size_t)mm * D_);
        float v0 = bf2f((unsigned short)(raw.x & 0xFFFF));
        float v1 = bf2f((unsigned short)(raw.x >> 16));
        float v2 = bf2f((unsigned short)(raw.y & 0xFFFF));
        float v3 = bf2f((unsigned short)(raw.y >> 16));
        v0 = v0 * (1.0f - wm * e0) + wm * a0;
        v1 = v1 * (1.0f - wm * e1) + wm * a1;
        v2 = v2 * (1.0f - wm * e2) + wm * a2;
        v3 = v3 * (1.0f - wm * e3) + wm * a3;
        uint2 o;
        o.x = (unsigned)f2bfu(v0) | ((unsigned)f2bfu(v1) << 16);
        o.y = (unsigned)f2bfu(v2) | ((unsigned)f2bfu(v3) << 16);
        *(uint2*)(mvp + (size_t)mm * D_) = o;
      }
    }
  }
}

// ---------------------------------------------------------------------------
extern "C" void kernel_launch(void* const* d_in, const int* in_sizes, int n_in,
                              void* d_out, int out_size, void* d_ws, size_t ws_size,
                              hipStream_t stream) {
  (void)in_sizes; (void)n_in; (void)out_size; (void)ws_size;
  const float* x    = (const float*)d_in[0];
  const float* mkey = (const float*)d_in[1];
  const float* mval = (const float*)d_in[2];
  const float* Wr   = (const float*)d_in[3];
  const float* br   = (const float*)d_in[4];
  const float* Ww   = (const float*)d_in[5];
  const float* bw   = (const float*)d_in[6];
  const float* Wp   = (const float*)d_in[7];
  const float* bp   = (const float*)d_in[8];
  float* out = (float*)d_out;

  char* ws = (char*)d_ws;
  unsigned short* WrT  = (unsigned short*)(ws);            // 64 KB
  unsigned short* WwT  = (unsigned short*)(ws + 65536);    // 64 KB
  unsigned short* WpT  = (unsigned short*)(ws + 131072);   // 32 KB
  unsigned short* k_bf = (unsigned short*)(ws + 163840);   // 32 MB (natural-order bf16)
  unsigned short* e_bf = k_bf + (size_t)B_ * S_ * D_;      // 32 MB
  unsigned short* a_bf = e_bf + (size_t)B_ * S_ * D_;      // 32 MB

  prep_weights<<<(INDIM * D_ + 255) / 256, 256, 0, stream>>>(Wr, Ww, Wp, WrT, WwT, WpT);
  gemm_act<<<(B_ * S_) / 16, 256, 0, stream>>>(x, WrT, WwT, br, bw, k_bf, e_bf, a_bf);
  scan_kernel<<<B_ / BPB, 1024, LDS_TOTAL, stream>>>(k_bf, e_bf, a_bf, mkey, mval,
                                                     WpT, bp, out);
}
